// MultiHeadAttention_19456201850932
// MI455X (gfx1250) — matrix operand order repaired and checked by reading, not compiled, not run on hardware
//
#include <hip/hip_runtime.h>

#define D_MODEL   1024
#define NUM_HEADS 16
#define DK        64
#define BATCH     2
#define SEQ       2048
#define MROWS     (BATCH * SEQ)   // 4096

typedef __attribute__((ext_vector_type(16))) __bf16 v16bf;
typedef __attribute__((ext_vector_type(8)))  __bf16 v8bf;
typedef __attribute__((ext_vector_type(8)))  float  v8f;

__device__ __forceinline__ v8f wmma_bf16(v16bf a, v16bf b, v8f c) {
  // (neg_a, A, neg_b, B, c_mod, C, reuse_a, reuse_b)
  return __builtin_amdgcn_wmma_f32_16x16x32_bf16(false, a, false, b, (short)0, c,
                                                 false, false);
}

// ---- CDNA5 async global->LDS copy (ASYNCcnt-tracked, ISA 15.18.3 op 98) ----
// vdst = per-lane LDS byte address, GVS mode: addr = saddr + vaddr32.
__device__ __forceinline__ void async_load_b128(unsigned lds_off,
                                                const void* sbase,
                                                unsigned voff) {
  asm volatile("global_load_async_to_lds_b128 %0, %1, %2 offset:0"
               :: "v"(lds_off), "v"(voff), "s"(sbase)
               : "memory");
}
__device__ __forceinline__ void wait_async_le2() {
  asm volatile("s_wait_asynccnt 2" ::: "memory");
}
__device__ __forceinline__ void wait_async_le0() {
  asm volatile("s_wait_asynccnt 0" ::: "memory");
}
// Low 32 bits of a flat LDS pointer are the LDS byte offset (aperture rule).
__device__ __forceinline__ unsigned lds_offset(const void* p) {
  return (unsigned)(unsigned long long)p;
}

// ---- fragment loaders (wave32, contiguous k = 16*hi + s, same order for A and B) -------------------------
// A (16x32, 16-bit): lane row = lane&15; lanes 0-15 hold K=0..15,
// lanes 16-31 hold K=16..31 (contiguous per lane). p = &row[k0].
__device__ __forceinline__ v16bf a_frag_f32(const float* p, int hi) {
  v16bf f;
#pragma unroll
  for (int i = 0; i < 8; ++i) f[i] = (__bf16)p[hi * 16 + i];
#pragma unroll
  for (int i = 0; i < 8; ++i) f[8 + i] = (__bf16)p[hi * 16 + 8 + i];
  return f;
}
__device__ __forceinline__ v16bf a_frag_bf16(const __bf16* p, int hi) {
  v8bf lo = *(const v8bf*)(p + hi * 16);
  v8bf hh = *(const v8bf*)(p + hi * 16 + 8);
  v16bf f;
#pragma unroll
  for (int i = 0; i < 8; ++i) { f[i] = lo[i]; f[8 + i] = hh[i]; }
  return f;
}
// B (32x16, 16-bit): lane col = lane&15; lanes 0-15 hold K=0..15,
// lanes 16-31 hold K=16..31 (contiguous per lane). p = &col[k0].
__device__ __forceinline__ v16bf b_frag_f32(const float* p, int hi) {
  v16bf f;
#pragma unroll
  for (int i = 0; i < 16; ++i) f[i] = (__bf16)p[hi * 16 + i];
  return f;
}
__device__ __forceinline__ v16bf b_frag_bf16(const __bf16* p, int hi) {
  return *(const v16bf*)(p + hi * 16);  // 32B aligned contiguous
}
__device__ __forceinline__ v16bf b_frag_bf16_strided(const __bf16* p, int hi,
                                                     int stride) {
  v16bf f;
#pragma unroll
  for (int i = 0; i < 16; ++i) f[i] = p[(size_t)(hi * 16 + i) * stride];
  return f;
}

// ---- projection GEMM: out[m,n] = sum_k A[m,k] * W[n,k] + bias[n] ----------
// block = 256 thr (8 waves), block tile 128(M)x64(N), wave tile 16x64.
// The 64x32 f32 W tile is staged once per block into LDS with
// GLOBAL_LOAD_ASYNC_TO_LDS_B128 (double-buffered), killing the 8x per-wave
// redundancy and overlapping the next tile's fetch with this tile's WMMAs.
// OUT_MODE 0: f32 row-major [M, D_MODEL]
// OUT_MODE 1: bf16 split-head [B, H, S, DK]
#define WROW 36  // padded f32 row stride (144B): conflict-free, 16B aligned

template <bool A_BF16, int OUT_MODE>
__global__ __launch_bounds__(256) void proj_gemm(const void* __restrict__ Ap,
                                                 const float* __restrict__ W,
                                                 const float* __restrict__ bias,
                                                 void* __restrict__ outp) {
  __shared__ float wt[2][64 * WROW];

  const int tid  = threadIdx.x;
  const int wv   = tid >> 5;
  const int lane = tid & 31;
  const int r    = lane & 15;
  const int hi   = lane >> 4;
  const int m0   = blockIdx.x * 128 + wv * 16;
  const int n0   = blockIdx.y * 64;

  // Stage one 64(N) x 32(K) f32 W tile: 512 16-byte chunks, 2 per thread.
  auto stage = [&](int k0, int buf) {
#pragma unroll
    for (int j = 0; j < 2; ++j) {
      const int linear = j * 256 + tid;
      const int row = linear >> 3;  // 0..63  (N within tile)
      const int ck  = linear & 7;   // 16B chunk within the 32-float row
      const unsigned voff =
          (unsigned)(((n0 + row) * D_MODEL + k0 + ck * 4) * 4);
      const unsigned loff = lds_offset(&wt[buf][row * WROW + ck * 4]);
      async_load_b128(loff, W, voff);
    }
  };

  stage(0, 0);

  v8f acc[4] = {};
  for (int k0 = 0; k0 < D_MODEL; k0 += 32) {
    const int buf   = (k0 >> 5) & 1;
    const bool more = (k0 + 32) < D_MODEL;
    if (more) stage(k0 + 32, buf ^ 1);
    if (more) wait_async_le2(); else wait_async_le0();
    __syncthreads();

    v16bf a;
    if constexpr (A_BF16) {
      const __bf16* ap = (const __bf16*)Ap + (size_t)(m0 + r) * D_MODEL + k0;
      a = a_frag_bf16(ap, hi);
      if (more) __builtin_prefetch(ap + 32, 0, 1);  // -> global_prefetch_b8
    } else {
      const float* ap = (const float*)Ap + (size_t)(m0 + r) * D_MODEL + k0;
      a = a_frag_f32(ap, hi);
      if (more) __builtin_prefetch(ap + 32, 0, 1);
    }
#pragma unroll
    for (int j = 0; j < 4; ++j) {
      v16bf b = b_frag_f32(&wt[buf][(j * 16 + r) * WROW], hi);
      acc[j] = wmma_bf16(a, b, acc[j]);
    }
    __syncthreads();  // reads of wt[buf] done before it is restaged
  }

#pragma unroll
  for (int j = 0; j < 4; ++j) {
#pragma unroll
    for (int i = 0; i < 8; ++i) {
      const int m = m0 + hi * 8 + i;
      const int n = n0 + j * 16 + r;
      const float v = acc[j][i] + bias[n];
      if constexpr (OUT_MODE == 0) {
        ((float*)outp)[(size_t)m * D_MODEL + n] = v;
      } else {
        const int bb = m >> 11, s = m & (SEQ - 1);
        const int h  = n >> 6,  d = n & (DK - 1);
        ((__bf16*)outp)[((size_t)(bb * NUM_HEADS + h) * SEQ + s) * DK + d] =
            (__bf16)v;
      }
    }
  }
}

// ---- fused attention: one WG per (b, h, 32-row q tile) ---------------------
// 32 x 2048 f32 score tile kept entirely in LDS (256 KB of the 320 KB WGP LDS).
// attn probs leave the chip exactly once (the roofline-dominant 537 MB).
__global__ __launch_bounds__(256) void attn_fused(
    const __bf16* __restrict__ q, const __bf16* __restrict__ k,
    const __bf16* __restrict__ v, float* __restrict__ attn_out,
    __bf16* __restrict__ ctx) {
  extern __shared__ float sc[];  // [32][SEQ]

  const int tid  = threadIdx.x;
  const int wv   = tid >> 5;
  const int lane = tid & 31;
  const int r    = lane & 15;
  const int hi   = lane >> 4;

  const int qbase = blockIdx.x * 32;
  const int h     = blockIdx.y;
  const int b     = blockIdx.z;
  const int bh    = b * NUM_HEADS + h;

  const __bf16* qh = q + (size_t)bh * SEQ * DK;
  const __bf16* kh = k + (size_t)bh * SEQ * DK;
  const __bf16* vh = v + (size_t)bh * SEQ * DK;

  const int mh = wv >> 2;   // row half: rows [mh*16, mh*16+16)
  const int nq = wv & 3;    // col quarter of the 2048-wide key range
  const int m0 = mh * 16;

  // -------- phase 1: scores = (q k^T) * scale, causal mask, into LDS ------
  v16bf aq[2];
#pragma unroll
  for (int ks = 0; ks < 2; ++ks)
    aq[ks] = a_frag_bf16(qh + (size_t)(qbase + m0 + r) * DK + ks * 32, hi);

  const float scale = 0.125f;  // 1/sqrt(64)
  for (int nt = 0; nt < 32; ++nt) {
    const int n0 = nq * 512 + nt * 16;
    v8f acc = {};
#pragma unroll
    for (int ks = 0; ks < 2; ++ks) {
      v16bf bk = b_frag_bf16(kh + (size_t)(n0 + r) * DK + ks * 32, hi);
      acc = wmma_bf16(aq[ks], bk, acc);
    }
    const int kg = n0 + r;
#pragma unroll
    for (int i = 0; i < 8; ++i) {
      const int lm = m0 + hi * 8 + i;
      const int qg = qbase + lm;
      float s = acc[i] * scale;
      if (kg > qg) s = -__builtin_inff();
      sc[lm * SEQ + kg] = s;
    }
  }
  __syncthreads();

  // -------- phase 2: softmax per row (one wave per row, wave32 shuffles) --
  for (int row = wv; row < 32; row += 8) {
    float* rp = sc + row * SEQ;
    float mx = -__builtin_inff();
    for (int c = lane; c < SEQ; c += 32) mx = fmaxf(mx, rp[c]);
#pragma unroll
    for (int o = 16; o > 0; o >>= 1) mx = fmaxf(mx, __shfl_xor(mx, o, 32));
    float sum = 0.f;
    for (int c = lane; c < SEQ; c += 32) {
      const float e = __expf(rp[c] - mx);
      rp[c] = e;
      sum += e;
    }
#pragma unroll
    for (int o = 16; o > 0; o >>= 1) sum += __shfl_xor(sum, o, 32);
    const float inv = 1.0f / sum;
    for (int c = lane; c < SEQ; c += 32) rp[c] *= inv;
  }
  __syncthreads();

  // -------- phase 3: stream probs to the attn output (single write) ------
  {
    const float4* src = (const float4*)sc;
    float4* dst = (float4*)(attn_out + ((size_t)bh * SEQ + qbase) * SEQ);
    for (int idx = tid; idx < 32 * SEQ / 4; idx += 256) dst[idx] = src[idx];
  }

  // -------- phase 4: ctx = P @ V (A from LDS, B bf16 from global/L2) ------
  v8f acc = {};
  for (int kt = 0; kt < SEQ / 32; ++kt) {
    v16bf a = a_frag_f32(sc + (size_t)(m0 + r) * SEQ + kt * 32, hi);
    v16bf bv =
        b_frag_bf16_strided(vh + (size_t)(kt * 32) * DK + nq * 16 + r, hi, DK);
    acc = wmma_bf16(a, bv, acc);
  }
#pragma unroll
  for (int i = 0; i < 8; ++i) {
    const int lm = m0 + hi * 8 + i;
    const int gq = qbase + lm;
    ctx[(size_t)(b * SEQ + gq) * D_MODEL + h * DK + nq * 16 + r] =
        (__bf16)acc[i];
  }
}

extern "C" void kernel_launch(void* const* d_in, const int* in_sizes, int n_in,
                              void* d_out, int out_size, void* d_ws,
                              size_t ws_size, hipStream_t stream) {
  const float* Q   = (const float*)d_in[0];
  const float* K   = (const float*)d_in[1];
  const float* V   = (const float*)d_in[2];
  // d_in[3] = causal mask (applied analytically)
  const float* W_Q = (const float*)d_in[4];
  const float* b_Q = (const float*)d_in[5];
  const float* W_K = (const float*)d_in[6];
  const float* b_K = (const float*)d_in[7];
  const float* W_V = (const float*)d_in[8];
  const float* b_V = (const float*)d_in[9];
  const float* W_O = (const float*)d_in[10];
  const float* b_O = (const float*)d_in[11];

  const size_t head_elems = (size_t)BATCH * NUM_HEADS * SEQ * DK;  // 4M
  __bf16* qw  = (__bf16*)d_ws;
  __bf16* kw  = qw + head_elems;
  __bf16* vw  = kw + head_elems;
  __bf16* ctx = vw + head_elems;  // [B*S, D_MODEL] bf16

  float* out  = (float*)d_out;                  // [B,S,D_MODEL]
  float* attn = out + (size_t)MROWS * D_MODEL;  // [B,H,S,S]

  const dim3 gp(MROWS / 128, D_MODEL / 64);
  proj_gemm<false, 1><<<gp, 256, 0, stream>>>(Q, W_Q, b_Q, qw);
  proj_gemm<false, 1><<<gp, 256, 0, stream>>>(K, W_K, b_K, kw);
  proj_gemm<false, 1><<<gp, 256, 0, stream>>>(V, W_V, b_V, vw);

  const int lds_bytes = 32 * SEQ * (int)sizeof(float);  // 256 KB
  (void)hipFuncSetAttribute((const void*)attn_fused,
                            hipFuncAttributeMaxDynamicSharedMemorySize,
                            lds_bytes);
  const dim3 ga(SEQ / 32, NUM_HEADS, BATCH);
  attn_fused<<<ga, 256, lds_bytes, stream>>>(qw, kw, vw, attn, ctx);

  proj_gemm<true, 0><<<gp, 256, 0, stream>>>(ctx, W_O, b_O, out);
}
